// EdgeEditor_78975858639699
// MI455X (gfx1250) — matrix operand order repaired and checked by reading, not compiled, not run on hardware
//
#include <hip/hip_runtime.h>
#include <stdint.h>

typedef __attribute__((ext_vector_type(16))) __bf16 v16bf;
typedef __attribute__((ext_vector_type(8)))  __bf16 v8bf;
typedef __attribute__((ext_vector_type(8)))  float  v8f;

namespace {

constexpr int kB   = 16;
constexpr int kS   = 2048;
constexpr int kN   = 256;
constexpr int kE   = 4096;
constexpr int kH   = 512;
constexpr int kLD  = 16;
constexpr int kETR = 768;
constexpr int kD   = 528;   // H + LD
constexpr int kC1  = 3;
constexpr int kGL  = 2;
constexpr int kNN  = 257;   // N + 1 graph nodes

__device__ __forceinline__ __bf16 f2bf(float f) {
  unsigned u = __float_as_uint(f);
  u += 0x7FFFu + ((u >> 16) & 1u);          // round-to-nearest-even
  unsigned short s = (unsigned short)(u >> 16);
  __bf16 r;
  __builtin_memcpy(&r, &s, sizeof(r));
  return r;
}

// Monotonic float -> uint mapping so uint atomicMax == float max.
__device__ __forceinline__ unsigned f_ord(float f) {
  unsigned u = __float_as_uint(f);
  return (u & 0x80000000u) ? ~u : (u | 0x80000000u);
}

__device__ __forceinline__ v16bf cat16(v8bf lo, v8bf hi) {
  return __builtin_shufflevector(lo, hi, 0,1,2,3,4,5,6,7,8,9,10,11,12,13,14,15);
}

// Stage one 64x32 A tile and one 32x128 B tile (fp32 -> bf16) into LDS.
// Fast paths use vector loads / packed b128 stores; edge paths are branchless
// (clamped unconditional loads + select-to-zero, no exec-mask divergence).
__device__ __forceinline__ void stage_tile(
    __bf16 (*As)[40], __bf16 (*Bf)[32][24],
    const float* __restrict__ Ab, int lda,
    const float* __restrict__ Bw, int ldb,
    int m0, int n0, int k0, int M, int N, int K,
    int ar, int ac, int bc, int bq, int bg, int bl,
    bool tileM, bool tileN)
{
  const bool kFull = (k0 + 32 <= K);

  // ---- A ----
  if (tileM & kFull) {
    const float* base = Ab + (long)(m0 + ar) * lda + k0 + ac;
    if (k0 + 96 <= K) __builtin_prefetch(base + 64, 0, 0);   // 2 K-steps ahead
    float4 x0 = *(const float4*)(base);
    float4 x1 = *(const float4*)(base + 4);
    v8bf pk = { f2bf(x0.x), f2bf(x0.y), f2bf(x0.z), f2bf(x0.w),
                f2bf(x1.x), f2bf(x1.y), f2bf(x1.z), f2bf(x1.w) };
    *(v8bf*)&As[ar][ac] = pk;
  } else {
    const int gm  = m0 + ar;
    const int gmc = gm < M ? gm : (M - 1);
    v8bf pk;
    #pragma unroll
    for (int j = 0; j < 8; ++j) {
      int gk  = k0 + ac + j;
      int gkc = gk < K ? gk : (K - 1);
      float v = Ab[(long)gmc * lda + gkc];          // always legal
      pk[j] = f2bf((gm < M && gk < K) ? v : 0.0f);  // cndmask, no branch
    }
    *(v8bf*)&As[ar][ac] = pk;
  }

  // ---- B (swizzled to fragment-major layout) ----
  if (tileN & kFull) {
    const float* bp = Bw + (long)(k0 + bq * 16) * ldb + n0 + bc;
    v8bf p0, p1;
    #pragma unroll
    for (int j = 0; j < 8; ++j) p0[j] = f2bf(bp[(long)j * ldb]);
    #pragma unroll
    for (int j = 0; j < 8; ++j) p1[j] = f2bf(bp[(long)(8 + j) * ldb]);
    *(v8bf*)&Bf[bg][bl][0] = p0;
    *(v8bf*)&Bf[bg][bl][8] = p1;
  } else {
    const int gn  = n0 + bc;
    const int gnc = gn < N ? gn : (N - 1);
    v8bf p0, p1;
    #pragma unroll
    for (int j = 0; j < 16; ++j) {
      int gk  = k0 + bq * 16 + j;
      int gkc = gk < K ? gk : (K - 1);
      float v = Bw[(long)gkc * ldb + gnc];
      __bf16 h = f2bf((gk < K && gn < N) ? v : 0.0f);
      if (j < 8) p0[j] = h; else p1[j - 8] = h;
    }
    *(v8bf*)&Bf[bg][bl][0] = p0;
    *(v8bf*)&Bf[bg][bl][8] = p1;
  }
}

// ---------------------------------------------------------------------------
// bf16-WMMA GEMM:  C[z] = A[z] @ Bw (+ bias), A row-major MxK, Bw row-major
// KxN.  Block tile 64(M) x 128(N), 256 threads = 8 waves; each wave owns a
// 32x32 sub-tile (2 A-frags x 2 B-frags = 4 v_wmma per K-step of 32).
// Ping-pong LDS double buffering: stage K-step k+32 while computing k, one
// barrier per iteration, global loads overlap the WMMAs.
// Optional fused segment-max epilogue (seg_out != null): atomicMax into
// seg_out[(row/kS)*kNN + seg_idx[row]][n] with order-preserving encoding.
// ---------------------------------------------------------------------------
__global__ __launch_bounds__(256) void wmma_gemm(
    const float* __restrict__ A, int lda, long strideA,
    const float* __restrict__ Bw, int ldb,
    const float* __restrict__ bias,
    float* __restrict__ C, int ldc, long strideC,
    int M, int N, int K,
    const int* __restrict__ seg_idx, unsigned* __restrict__ seg_out)
{
  __shared__ __bf16 As[2][64][40];       // 2 x 5.0 KB
  __shared__ __bf16 Bf[2][8][32][24];    // 2 x 12.0 KB

  const int tid  = threadIdx.x;
  const int wave = tid >> 5;
  const int lane = tid & 31;
  const int l15  = lane & 15;
  const int lhf  = lane >> 4;

  const int m0 = blockIdx.x * 64;
  const int n0 = blockIdx.y * 128;
  const float* Ab = A + (long)blockIdx.z * strideA;
  float*       Cb = C ? (C + (long)blockIdx.z * strideC) : nullptr;

  const int wm = (wave >> 2) * 32;    // 2 waves along M (rows wm..wm+31)
  const int wn = (wave & 3) * 32;     // 4 waves along N (cols wn..wn+31)
  const int g0 = wn >> 4;             // B fragment groups g0, g0+1

  // Staging assignments (fixed per thread).
  const int ar = tid >> 2;            // A: row 0..63
  const int ac = (tid & 3) * 8;       // A: k-chunk 0/8/16/24
  const int bc = tid & 127;           // B: column 0..127
  const int bq = tid >> 7;            // B: k-half 0/1 (k = bq*16 + j)
  const int bg = bc >> 4;             // B: n-group
  const int bl = (bc & 15) + 16 * bq; // B: fragment lane slot

  const bool tileM = (m0 + 64 <= M);
  const bool tileN = (n0 + 128 <= N);

  v8f acc00 = {0.f,0.f,0.f,0.f,0.f,0.f,0.f,0.f};
  v8f acc01 = acc00, acc10 = acc00, acc11 = acc00;

  // Prologue: stage first K-step into buffer 0.
  stage_tile(As[0], Bf[0], Ab, lda, Bw, ldb, m0, n0, 0, M, N, K,
             ar, ac, bc, bq, bg, bl, tileM, tileN);
  __syncthreads();

  int cur = 0;
  for (int k0 = 0; k0 < K; k0 += 32) {
    // Stage next K-step into the other buffer (overlaps WMMAs below).
    if (k0 + 32 < K)
      stage_tile(As[cur ^ 1], Bf[cur ^ 1], Ab, lda, Bw, ldb, m0, n0, k0 + 32,
                 M, N, K, ar, ac, bc, bq, bg, bl, tileM, tileN);

    // ---- fragments: all contiguous b128 LDS loads ----
    const int kb = 16 * lhf;          // A: lanes16-31 hold K+16 offset
    const int mA0 = wm + l15;
    const int mA1 = wm + 16 + l15;
    v16bf a0 = cat16(*(const v8bf*)&As[cur][mA0][kb], *(const v8bf*)&As[cur][mA0][kb + 8]);
    v16bf a1 = cat16(*(const v8bf*)&As[cur][mA1][kb], *(const v8bf*)&As[cur][mA1][kb + 8]);
    v16bf b0 = cat16(*(const v8bf*)&Bf[cur][g0][lane][0],     *(const v8bf*)&Bf[cur][g0][lane][8]);
    v16bf b1 = cat16(*(const v8bf*)&Bf[cur][g0 + 1][lane][0], *(const v8bf*)&Bf[cur][g0 + 1][lane][8]);

    acc00 = __builtin_amdgcn_wmma_f32_16x16x32_bf16(false, a0, false, b0, (short)0, acc00, false, false);
    acc01 = __builtin_amdgcn_wmma_f32_16x16x32_bf16(false, a0, false, b1, (short)0, acc01, false, false);
    acc10 = __builtin_amdgcn_wmma_f32_16x16x32_bf16(false, a1, false, b0, (short)0, acc10, false, false);
    acc11 = __builtin_amdgcn_wmma_f32_16x16x32_bf16(false, a1, false, b1, (short)0, acc11, false, false);
    __syncthreads();
    cur ^= 1;
  }

  // ---- epilogue ----
  // D layout: VGPR r -> M=r (lanes0-15) / M=8+r (lanes16-31), N = lane&15.
  #pragma unroll
  for (int s = 0; s < 2; ++s) {
    #pragma unroll
    for (int t = 0; t < 2; ++t) {
      v8f acc = s ? (t ? acc11 : acc10) : (t ? acc01 : acc00);
      int gn = n0 + wn + t * 16 + l15;
      if (gn >= N) continue;
      float bv = bias ? bias[gn] : 0.0f;
      #pragma unroll
      for (int r = 0; r < 8; ++r) {
        int gm = m0 + wm + s * 16 + lhf * 8 + r;
        if (gm >= M) continue;
        float v = acc[r] + bv;
        if (seg_out) {
          int b    = gm / kS;
          int node = seg_idx[gm];
          atomicMax(&seg_out[((long)b * kNN + node) * N + gn], f_ord(v));
        } else {
          Cb[(long)gm * ldc + gn] = v;
        }
      }
    }
  }
}

// ---------------------------------------------------------------------------
__global__ void fill_u32(unsigned* __restrict__ p, long n, unsigned v) {
  long t = blockIdx.x * (long)blockDim.x + threadIdx.x;
  if (t < n) p[t] = v;
}

__global__ void scale_f32(float* __restrict__ p, long n, float s) {
  long t = blockIdx.x * (long)blockDim.x + threadIdx.x;
  if (t < n) p[t] *= s;
}

// x0 = concat(label_part, relu(segmax)) ; node 0: [tbl[0], zeros]
__global__ void build_x0(const unsigned* __restrict__ seg,
                         const int* __restrict__ ent_labels,
                         const float* __restrict__ tbl,
                         float* __restrict__ x0)
{
  int bn = blockIdx.x;              // b*kNN + n
  int b = bn / kNN, n = bn % kNN;
  for (int d = threadIdx.x; d < kD; d += blockDim.x) {
    float v;
    if (d < kLD) {
      if (n == 0) v = tbl[d];
      else {
        int l = ent_labels[b * kN + (n - 1)];
        v = (l != 0) ? tbl[l * kLD + d] : 0.0f;
      }
    } else {
      if (n == 0) v = 0.0f;
      else {
        unsigned e = seg[((long)b * kNN + n) * kH + (d - kLD)];
        // decode + ReLU: nonnegative iff ordered-code MSB set
        v = (e & 0x80000000u) ? __uint_as_float(e & 0x7FFFFFFFu) : 0.0f;
      }
    }
    x0[(long)bn * kD + d] = v;
  }
}

__global__ void count_deg(const int* __restrict__ edge_index,
                          const int* __restrict__ edge_attr,
                          int* __restrict__ degcnt)
{
  int t = blockIdx.x * blockDim.x + threadIdx.x;   // over B*E
  if (t >= kB * kE) return;
  int b = t / kE, e = t % kE;
  int c = edge_attr[t];
  int dst = edge_index[b * 2 * kE + kE + e];
  if (c >= 0 && c < kC1) atomicAdd(&degcnt[((long)c * kB + b) * kNN + dst], 1);
}

__global__ void make_dinv(const int* __restrict__ degcnt, float* __restrict__ dinv) {
  int t = blockIdx.x * blockDim.x + threadIdx.x;
  if (t < kC1 * kB * kNN) dinv[t] = rsqrtf(1.0f + (float)degcnt[t]);
}

// agg[dst] += dinv[src]*dinv[dst] * h[src]   for edges of class `cls`
__global__ void edge_agg(const int* __restrict__ edge_index,
                         const int* __restrict__ edge_attr, int cls,
                         const float* __restrict__ dinv_c,
                         const float* __restrict__ h,
                         float* __restrict__ agg)
{
  int t = blockIdx.x;               // b*kE + e
  int b = t / kE, e = t % kE;
  if (edge_attr[t] != cls) return;
  int src = edge_index[b * 2 * kE + e];
  int dst = edge_index[b * 2 * kE + kE + e];
  float coef = dinv_c[b * kNN + src] * dinv_c[b * kNN + dst];
  const float* hs = h   + ((long)b * kNN + src) * kD;
  float*       ad = agg + ((long)b * kNN + dst) * kD;
  for (int d = threadIdx.x; d < kD; d += blockDim.x)
    atomicAdd(&ad[d], coef * hs[d]);
}

// out = agg + h/deg + bias   (deg = 1/dinv^2); accumulate-mode adds into out
__global__ void gcn_combine(const float* __restrict__ agg,
                            const float* __restrict__ h,
                            const float* __restrict__ dinv_c,
                            const float* __restrict__ bias,
                            float* __restrict__ out, int accumulate)
{
  int bn = blockIdx.x;
  float di = dinv_c[bn];
  float invdeg = di * di;
  for (int d = threadIdx.x; d < kD; d += blockDim.x) {
    long idx = (long)bn * kD + d;
    float v = agg[idx] + h[idx] * invdeg + bias[d];
    if (accumulate) out[idx] += v; else out[idx] = v;
  }
}

// Y[b,r,c] = X[b, node_off + r, :] . W_out[:, c]   (C1 = 3 columns)
__global__ void small_out(const float* __restrict__ X, int rows, int node_off,
                          const float* __restrict__ Wo, float* __restrict__ Y)
{
  int t = blockIdx.x * blockDim.x + threadIdx.x;
  int total = kB * rows * kC1;
  if (t >= total) return;
  int c = t % kC1;
  int r = (t / kC1) % rows;
  int b = t / (kC1 * rows);
  const float* x = X + ((long)b * kNN + node_off + r) * kD;
  float s = 0.0f;
  for (int d = 0; d < kD; ++d) s += x[d] * Wo[d * kC1 + c];
  Y[t] = s;
}

// logits[b,i,j,c] = (hW[b,i,c] + tW[b,j,c]) * cands[b,i,1+j] + b_out[c]
__global__ void logits_k(const float* __restrict__ hW,
                         const float* __restrict__ tW,
                         const unsigned char* __restrict__ cands,
                         const float* __restrict__ b_out,
                         float* __restrict__ out)
{
  long t = blockIdx.x * (long)blockDim.x + threadIdx.x;
  long total = (long)kB * kNN * kN * kC1;
  if (t >= total) return;
  int c = (int)(t % kC1);
  long r = t / kC1;
  int j = (int)(r % kN); r /= kN;
  int i = (int)(r % kNN);
  int b = (int)(r / kNN);
  float m = cands[((long)b * kNN + i) * kNN + (1 + j)] ? 1.0f : 0.0f;
  out[t] = (hW[((long)b * kNN + i) * kC1 + c] + tW[((long)b * kN + j) * kC1 + c]) * m + b_out[c];
}

} // namespace

extern "C" void kernel_launch(void* const* d_in, const int* in_sizes, int n_in,
                              void* d_out, int out_size, void* d_ws, size_t ws_size,
                              hipStream_t stream)
{
  (void)in_sizes; (void)n_in; (void)out_size; (void)ws_size;

  const float* emb        = (const float*)d_in[0];
  const int*   ent_idx    = (const int*)d_in[1];
  const int*   ent_labels = (const int*)d_in[2];
  const int*   edge_index = (const int*)d_in[3];
  const int*   edge_attr  = (const int*)d_in[4];
  const unsigned char* cands = (const unsigned char*)d_in[5];
  const float* W_red  = (const float*)d_in[6];
  const float* b_red  = (const float*)d_in[7];
  const float* tbl    = (const float*)d_in[8];
  const float* W_gcn  = (const float*)d_in[9];
  const float* b_gcn  = (const float*)d_in[10];
  const float* W_head = (const float*)d_in[11];
  const float* b_head = (const float*)d_in[12];
  const float* W_tail = (const float*)d_in[13];
  const float* b_tail = (const float*)d_in[14];
  const float* W_out  = (const float*)d_in[15];
  const float* b_out  = (const float*)d_in[16];
  float* out = (float*)d_out;

  // Carve workspace (4-byte elements, 256B aligned regions).
  size_t off = 0;
  auto carve = [&](size_t elems) -> void* {
    void* p = (char*)d_ws + off;
    off += elems * 4;
    off = (off + 255) & ~(size_t)255;
    return p;
  };
  const size_t nBND = (size_t)kB * kNN * kD;
  unsigned* seg    = (unsigned*)carve((size_t)kB * kNN * kH);
  float*    x0     = (float*)carve(nBND);
  float*    xbuf   = (float*)carve(nBND);
  float*    hbuf   = (float*)carve(nBND);
  float*    agg    = (float*)carve(nBND);
  float*    result = (float*)carve(nBND);
  float*    xhead  = (float*)carve(nBND);
  float*    xtail  = (float*)carve(nBND);
  int*      degcnt = (int*)carve((size_t)kC1 * kB * kNN);
  float*    dinv   = (float*)carve((size_t)kC1 * kB * kNN);
  float*    hW     = (float*)carve((size_t)kB * kNN * kC1);
  float*    tW     = (float*)carve((size_t)kB * kN * kC1);

  auto cdiv = [](long a, long b) { return (a + b - 1) / b; };

  // Init scratch that is read-modify-written.
  fill_u32<<<cdiv((long)kB * kNN * kH, 256), 256, 0, stream>>>(seg, (long)kB * kNN * kH, 0u);
  fill_u32<<<cdiv((long)kC1 * kB * kNN, 256), 256, 0, stream>>>((unsigned*)degcnt, (long)kC1 * kB * kNN, 0u);
  fill_u32<<<cdiv((long)nBND, 256), 256, 0, stream>>>((unsigned*)result, (long)nBND, 0u);

  // red = emb @ W_red + b_red, fused with per-batch segment-max scatter.
  {
    dim3 g((unsigned)cdiv((long)kB * kS, 64), (unsigned)cdiv(kH, 128), 1);
    wmma_gemm<<<g, 256, 0, stream>>>(emb, kETR, 0, W_red, kH, b_red,
                                     nullptr, 0, 0, kB * kS, kH, kETR,
                                     ent_idx, seg);
  }
  build_x0<<<kB * kNN, 256, 0, stream>>>(seg, ent_labels, tbl, x0);
  count_deg<<<cdiv((long)kB * kE, 256), 256, 0, stream>>>(edge_index, edge_attr, degcnt);
  make_dinv<<<cdiv((long)kC1 * kB * kNN, 256), 256, 0, stream>>>(degcnt, dinv);

  // GCN: 3 relation classes x 2 layers.
  for (int c = 0; c < kC1; ++c) {
    const float* dinv_c = dinv + (size_t)c * kB * kNN;
    for (int l = 0; l < kGL; ++l) {
      const float* xin = (l == 0) ? x0 : xbuf;
      const float* W   = W_gcn + (((size_t)c * kGL + l) * kD) * kD;
      const float* bb  = b_gcn + ((size_t)c * kGL + l) * kD;
      dim3 g((unsigned)cdiv(kNN, 64), (unsigned)cdiv(kD, 128), kB);
      wmma_gemm<<<g, 256, 0, stream>>>(xin, kD, (long)kNN * kD, W, kD, nullptr,
                                       hbuf, kD, (long)kNN * kD, kNN, kD, kD,
                                       nullptr, nullptr);
      fill_u32<<<cdiv((long)nBND, 256), 256, 0, stream>>>((unsigned*)agg, (long)nBND, 0u);
      edge_agg<<<kB * kE, 128, 0, stream>>>(edge_index, edge_attr, c, dinv_c, hbuf, agg);
      gcn_combine<<<kB * kNN, 256, 0, stream>>>(agg, hbuf, dinv_c, bb,
                                                (l == kGL - 1) ? result : xbuf,
                                                (l == kGL - 1) ? 1 : 0);
    }
  }
  scale_f32<<<cdiv((long)nBND, 256), 256, 0, stream>>>(result, (long)nBND, 1.0f / kC1);

  // Head / tail projections.
  {
    dim3 g((unsigned)cdiv((long)kB * kNN, 64), (unsigned)cdiv(kD, 128), 1);
    wmma_gemm<<<g, 256, 0, stream>>>(result, kD, 0, W_head, kD, b_head,
                                     xhead, kD, 0, kB * kNN, kD, kD, nullptr, nullptr);
    wmma_gemm<<<g, 256, 0, stream>>>(result, kD, 0, W_tail, kD, b_tail,
                                     xtail, kD, 0, kB * kNN, kD, kD, nullptr, nullptr);
  }
  small_out<<<cdiv((long)kB * kNN * kC1, 256), 256, 0, stream>>>(xhead, kNN, 0, W_out, hW);
  small_out<<<cdiv((long)kB * kN  * kC1, 256), 256, 0, stream>>>(xtail, kN, 1, W_out, tW);
  logits_k<<<cdiv((long)kB * kNN * kN * kC1, 256), 256, 0, stream>>>(hW, tW, cands, b_out, out);
}
